// NeighEnco2_61950608277608
// MI455X (gfx1250) — matrix operand order repaired and checked by reading, not compiled, not run on hardware
//
#include <hip/hip_runtime.h>
#include <hip/hip_bf16.h>

typedef __attribute__((ext_vector_type(16))) _Float16 v16h;
typedef __attribute__((ext_vector_type(4)))  _Float16 v4h;
typedef __attribute__((ext_vector_type(8)))  float    v8f;

#define C_IN   128
#define C_H    64
#define LNB    32
#define WAVES  4
#define NEG_SLOPE 0.05f

// ---- fragment loaders (LDS -> VGPR) ---------------------------------------
// A-matrix 16x32 f16 fragment (this code's contiguous k order, matched by B):
//   lane g = lane>>4, m = lane&15.  halfs 0..15 -> K = g*16 + 0..15
__device__ __forceinline__ v16h ld_frag_a(const _Float16* row, int kbase, int g) {
    union { v16h h; uint4 q[2]; } u;
    u.q[0] = *(const uint4*)(row + kbase + g * 16);
    u.q[1] = *(const uint4*)(row + kbase + g * 16 + 8);
    return u.h;
}
// B-matrix 32x16 f16 layout: lane j = lane&15 holds column j, K-group g=lane>>4,
// halfs 0..15 -> K = g*16 + 0..15 (contiguous in the row-major weight row).
__device__ __forceinline__ v16h ld_frag_b(const _Float16* wrow) {
    union { v16h h; uint4 q[2]; } u;
    u.q[0] = *(const uint4*)(wrow);
    u.q[1] = *(const uint4*)(wrow + 8);
    return u.h;
}

__device__ __forceinline__ v8f wmma_f16(v16h a, v16h b, v8f c) {
    return __builtin_amdgcn_wmma_f32_16x16x32_f16(false, a, false, b,
                                                  (short)0, c, false, false);
}

__global__ __launch_bounds__(WAVES * 32) void neigh_enco2_kernel(
    const float* __restrict__ z,            // [100000, 128]
    const long long* __restrict__ neighbor, // [n_total, 32]
    const float* __restrict__ w1,           // [64, 128]
    const float* __restrict__ w2,           // [64, 64]
    float* __restrict__ out,                // [n_total, 128]
    int n_total)
{
    __shared__ __align__(16) _Float16 s_w1[C_H][C_IN];           // 16 KB
    __shared__ __align__(16) _Float16 s_w2[C_H][C_H];            //  8 KB
    __shared__ __align__(16) _Float16 s_x [WAVES][LNB][C_IN];    // 32 KB (reused for h1)

    const int tid = threadIdx.x;

    // ---- stage weights fp32 -> f16 into LDS (once per block) --------------
    {
        _Float16* p1 = &s_w1[0][0];
        for (int i = tid; i < (C_H * C_IN) / 4; i += WAVES * 32) {
            float4 v = ((const float4*)w1)[i];
            v4h h; h.x = (_Float16)v.x; h.y = (_Float16)v.y;
                   h.z = (_Float16)v.z; h.w = (_Float16)v.w;
            *(v4h*)(p1 + i * 4) = h;
        }
        _Float16* p2 = &s_w2[0][0];
        for (int i = tid; i < (C_H * C_H) / 4; i += WAVES * 32) {
            float4 v = ((const float4*)w2)[i];
            v4h h; h.x = (_Float16)v.x; h.y = (_Float16)v.y;
                   h.z = (_Float16)v.z; h.w = (_Float16)v.w;
            *(v4h*)(p2 + i * 4) = h;
        }
    }
    __syncthreads();

    const int wave = tid >> 5;
    const int lane = tid & 31;
    const int n    = blockIdx.x * WAVES + wave;
    if (n >= n_total) return;

    const int g = lane >> 4;   // half-wave group
    const int j = lane & 15;   // column within tile

    // ---- gather: lane -> one neighbor row, fp32 -> f16 into LDS -----------
    {
        long long idx = neighbor[(long long)n * LNB + lane];
        const bool   zero = (idx == 0);
        const float* row  = z + (idx - 1) * (long long)C_IN;
        _Float16* dst = &s_x[wave][lane][0];
        #pragma unroll
        for (int c = 0; c < C_IN; c += 4) {
            float4 v = make_float4(0.f, 0.f, 0.f, 0.f);
            if (!zero) v = *(const float4*)(row + c);
            v4h h; h.x = (_Float16)v.x; h.y = (_Float16)v.y;
                   h.z = (_Float16)v.z; h.w = (_Float16)v.w;
            *(v4h*)(dst + c) = h;
        }
    }
    // per-wave LDS region; same-wave DS ops are in-order -> no block barrier.

    // ---- GEMM1: h1[32,64] = x[32,128] @ w1^T, 32 WMMAs --------------------
    v8f acc1[2][4];
    #pragma unroll
    for (int mt = 0; mt < 2; ++mt)
        #pragma unroll
        for (int nt = 0; nt < 4; ++nt) acc1[mt][nt] = (v8f){0,0,0,0,0,0,0,0};

    #pragma unroll
    for (int ks = 0; ks < 4; ++ks) {           // K = 128 in steps of 32
        v16h a0 = ld_frag_a(&s_x[wave][ 0 + j][0], ks * 32, g);
        v16h a1 = ld_frag_a(&s_x[wave][16 + j][0], ks * 32, g);
        #pragma unroll
        for (int nt = 0; nt < 4; ++nt) {
            v16h b = ld_frag_b(&s_w1[nt * 16 + j][ks * 32 + g * 16]);
            acc1[0][nt] = wmma_f16(a0, b, acc1[0][nt]);
            acc1[1][nt] = wmma_f16(a1, b, acc1[1][nt]);
        }
    }

    // ---- leaky-ReLU, column max, spill h1 (f16) into reused x region ------
    // C/D layout: VGPR r of lane -> row M = mt*16 + g*8 + r, col N = nt*16 + j
    _Float16* hx = &s_x[wave][0][0];           // rows of 64 halfs (first 4 KB)
    float cmax1[4] = {-3.4e38f, -3.4e38f, -3.4e38f, -3.4e38f};
    #pragma unroll
    for (int mt = 0; mt < 2; ++mt)
        #pragma unroll
        for (int nt = 0; nt < 4; ++nt) {
            v8f a = acc1[mt][nt];
            #pragma unroll
            for (int r = 0; r < 8; ++r) {
                float v = a[r];
                v = (v > 0.f) ? v : NEG_SLOPE * v;
                hx[(mt * 16 + g * 8 + r) * C_H + nt * 16 + j] = (_Float16)v;
                cmax1[nt] = fmaxf(cmax1[nt], v);
            }
        }

    // ---- GEMM2: h2[32,64] = h1[32,64] @ w2^T, 16 WMMAs --------------------
    v8f acc2[2][4];
    #pragma unroll
    for (int mt = 0; mt < 2; ++mt)
        #pragma unroll
        for (int nt = 0; nt < 4; ++nt) acc2[mt][nt] = (v8f){0,0,0,0,0,0,0,0};

    #pragma unroll
    for (int ks = 0; ks < 2; ++ks) {           // K = 64 in steps of 32
        v16h a0 = ld_frag_a(hx + ( 0 + j) * C_H, ks * 32, g);
        v16h a1 = ld_frag_a(hx + (16 + j) * C_H, ks * 32, g);
        #pragma unroll
        for (int nt = 0; nt < 4; ++nt) {
            v16h b = ld_frag_b(&s_w2[nt * 16 + j][ks * 32 + g * 16]);
            acc2[0][nt] = wmma_f16(a0, b, acc2[0][nt]);
            acc2[1][nt] = wmma_f16(a1, b, acc2[1][nt]);
        }
    }

    float cmax2[4] = {-3.4e38f, -3.4e38f, -3.4e38f, -3.4e38f};
    #pragma unroll
    for (int mt = 0; mt < 2; ++mt)
        #pragma unroll
        for (int nt = 0; nt < 4; ++nt) {
            v8f a = acc2[mt][nt];
            #pragma unroll
            for (int r = 0; r < 8; ++r) {
                float v = a[r];
                v = (v > 0.f) ? v : NEG_SLOPE * v;
                cmax2[nt] = fmaxf(cmax2[nt], v);
            }
        }

    // ---- combine half-waves (lanes L and L+16 hold the same column) -------
    float* orow = out + (long long)n * C_IN;
    #pragma unroll
    for (int nt = 0; nt < 4; ++nt) {
        float m1 = fmaxf(cmax1[nt], __shfl_xor(cmax1[nt], 16, 32));
        float m2 = fmaxf(cmax2[nt], __shfl_xor(cmax2[nt], 16, 32));
        if (g == 0) {
            orow[nt * 16 + j]        = m1;   // cat[:, :64]  = h1 max
            orow[64 + nt * 16 + j]   = m2;   // cat[:, 64:]  = h2 max
        }
    }
}

extern "C" void kernel_launch(void* const* d_in, const int* in_sizes, int n_in,
                              void* d_out, int out_size, void* d_ws, size_t ws_size,
                              hipStream_t stream) {
    const float*     z        = (const float*)d_in[0];
    const long long* neighbor = (const long long*)d_in[1];
    const float*     w1       = (const float*)d_in[2];
    const float*     w2       = (const float*)d_in[3];
    float*           out      = (float*)d_out;

    const int n_total = in_sizes[1] / LNB;                 // 50000
    const int blocks  = (n_total + WAVES - 1) / WAVES;     // 12500

    neigh_enco2_kernel<<<blocks, WAVES * 32, 0, stream>>>(
        z, neighbor, w1, w2, out, n_total);
}
